// RNNArgumentsHead_1357209666179
// MI455X (gfx1250) — matrix operand order repaired and checked by reading, not compiled, not run on hardware
//
#include <hip/hip_runtime.h>
#include <hip/hip_bf16.h>
#include <math.h>

typedef __bf16 bf16_t;
typedef __bf16 v16bf __attribute__((ext_vector_type(16)));
typedef __bf16 v8bf  __attribute__((ext_vector_type(8)));
typedef float  v8f   __attribute__((ext_vector_type(8)));

#define B_ 4096
#define T_ 16
#define H_ 512
#define E_ 256

// ---------------------------------------------------------------------------
// Hardware tanh (gfx1250 TRANS op) with fallbacks
// ---------------------------------------------------------------------------
__device__ __forceinline__ float fast_tanh(float x) {
#if defined(__HIP_DEVICE_COMPILE__)
#if __has_builtin(__builtin_amdgcn_tanhf)
    return __builtin_amdgcn_tanhf(x);
#elif __has_builtin(__builtin_amdgcn_tanh_f32)
    return __builtin_amdgcn_tanh_f32(x);
#else
    float r;
    asm("v_tanh_f32 %0, %1\n\ts_nop 0" : "=v"(r) : "v"(x));
    return r;
#endif
#else
    return tanhf(x);
#endif
}

// ---------------------------------------------------------------------------
// WMMA helpers (CDNA5 wave32, 16x16x32 bf16 -> f32)
// ---------------------------------------------------------------------------
__device__ __forceinline__ v8f wmma_bf16(v16bf a, v16bf b, v8f c) {
    return __builtin_amdgcn_wmma_f32_16x16x32_bf16(
        /*neg_a=*/false, a, /*neg_b=*/false, b,
        /*c_mod=*/(short)0, c, /*reuse_a=*/false, /*reuse_b=*/false);
}

// A fragment: 16x32 bf16 tile, row-major source with given row stride.
// lanes 0-15 (half=0): M=lane, K = k0+0..7 (VGPR0-3) and k0+16..23 (VGPR4-7)
// lanes 16-31 (half=1): M=lane-16, K = k0+8..15 and k0+24..31
__device__ __forceinline__ v16bf load_a_frag(const bf16_t* base, int stride,
                                             int m, int half, int k0) {
    union { v16bf v; v8bf h[2]; } u;
    const bf16_t* p = base + m * stride + k0 + half * 8;
    u.h[0] = *(const v8bf*)(p);
    u.h[1] = *(const v8bf*)(p + 16);
    return u.v;
}

// B fragment: 32x16 bf16 tile from a transposed weight Wt[N][K] (bf16).
// lane n=lane&15 selects column N=ncol; half=0 -> K=k0+0..7,k0+16..23; half=1 -> K=k0+8..15,k0+24..31.
__device__ __forceinline__ v16bf load_b_frag(const bf16_t* Wt, int K,
                                             int ncol, int half, int k0) {
    union { v16bf v; v8bf h[2]; } u;
    const bf16_t* p = Wt + (size_t)ncol * K + k0 + half * 8;
    u.h[0] = *(const v8bf*)(p);
    u.h[1] = *(const v8bf*)(p + 16);
    return u.v;
}

// ---------------------------------------------------------------------------
// Prep: f32 [K][N] -> bf16 transposed [N][K]
// ---------------------------------------------------------------------------
__global__ void transpose_to_bf16(const float* __restrict__ in,
                                  bf16_t* __restrict__ out, int K, int N) {
    int id = blockIdx.x * 256 + threadIdx.x;
    if (id < K * N) {
        int k = id / N, n = id % N;
        out[(size_t)n * K + k] = (bf16_t)in[id];
    }
}

// ---------------------------------------------------------------------------
// Layer 0: input is hidden_state broadcast over T, so the input projection
// xp = hs @ W0 + b0 is computed ONCE and held in registers. U-fragments are
// hoisted into registers for the whole scan; each serialized step is only
// ds_load A-fragments + 16 WMMAs + tanh.
// ---------------------------------------------------------------------------
__global__ __launch_bounds__(256, 1)
void rnn_layer0_kernel(const float* __restrict__ hs,     // [B][H] f32
                       const float* __restrict__ temb,   // [B][E] f32
                       const int*   __restrict__ nargs,  // [B]
                       const float* __restrict__ bias,   // [E] f32
                       const bf16_t* __restrict__ Wt,    // [E][H] bf16
                       const bf16_t* __restrict__ Ut,    // [E][E] bf16
                       bf16_t* __restrict__ seq,         // [B][T][E] bf16 (out)
                       bf16_t* __restrict__ hstate) {    // [B][E]    bf16 (out)
    __shared__ __align__(16) bf16_t hbuf[16 * E_];   // 8 KB recurrent state
    __shared__ __align__(16) bf16_t xbuf[16 * H_];   // 16 KB staged hs
    __shared__ int snargs[16];

    const int tid   = threadIdx.x;
    const int lane  = tid & 31;
    const int wave  = tid >> 5;
    const int half  = lane >> 4;
    const int n     = lane & 15;
    const int bbase = blockIdx.x * 16;

    if (tid < 16) snargs[tid] = nargs[bbase + tid];

    {
        const int row = tid >> 4;
        const int c16 = (tid & 15) * 16;
        const float* src = temb + (size_t)(bbase + row) * E_ + c16;
        #pragma unroll
        for (int i = 0; i < 16; ++i)
            hbuf[row * E_ + c16 + i] = (bf16_t)src[i];
        const int cc = (tid & 15) * 32;
        const float* s2 = hs + (size_t)(bbase + row) * H_ + cc;
        #pragma unroll
        for (int i = 0; i < 32; ++i)
            xbuf[row * H_ + cc + i] = (bf16_t)s2[i];
    }
    __syncthreads();

    const int c0 = wave * 32;
    const int c1 = c0 + 16;
    const float bv0 = bias[c0 + n];
    const float bv1 = bias[c1 + n];

    int na[8];
    #pragma unroll
    for (int r = 0; r < 8; ++r) na[r] = snargs[r + half * 8];

    v8f h0, h1;
    #pragma unroll
    for (int r = 0; r < 8; ++r) {
        int row = r + half * 8;
        h0[r] = (float)hbuf[row * E_ + c0 + n];
        h1[r] = (float)hbuf[row * E_ + c1 + n];
    }

    // hoist recurrent weight fragments for the whole scan (2*8*8 VGPRs)
    v16bf uf0[8], uf1[8];
    #pragma unroll
    for (int ks = 0; ks < 8; ++ks) {
        uf0[ks] = load_b_frag(Ut, E_, c0 + n, half, ks * 32);
        uf1[ks] = load_b_frag(Ut, E_, c1 + n, half, ks * 32);
    }

    // one-time input projection: xp = hs @ W0 + b0 (K = 512, W0 streamed)
    v8f xp0, xp1;
    #pragma unroll
    for (int r = 0; r < 8; ++r) { xp0[r] = bv0; xp1[r] = bv1; }
    #pragma unroll
    for (int ks = 0; ks < H_ / 32; ++ks) {
        int k0 = ks * 32;
        v16bf a  = load_a_frag(xbuf, H_, n, half, k0);
        v16bf w0 = load_b_frag(Wt, H_, c0 + n, half, k0);
        v16bf w1 = load_b_frag(Wt, H_, c1 + n, half, k0);
        xp0 = wmma_bf16(a, w0, xp0);
        xp1 = wmma_bf16(a, w1, xp1);
    }

    for (int t = 0; t < T_; ++t) {
        v8f acc0 = xp0, acc1 = xp1;
        #pragma unroll
        for (int ks = 0; ks < 8; ++ks) {
            v16bf a = load_a_frag(hbuf, E_, n, half, ks * 32);
            acc0 = wmma_bf16(a, uf0[ks], acc0);
            acc1 = wmma_bf16(a, uf1[ks], acc1);
        }
        #pragma unroll
        for (int r = 0; r < 8; ++r) {
            bool mt = t < na[r];
            h0[r] = mt ? fast_tanh(acc0[r]) : h0[r];
            h1[r] = mt ? fast_tanh(acc1[r]) : h1[r];
        }
        __syncthreads();
        #pragma unroll
        for (int r = 0; r < 8; ++r) {
            int row = r + half * 8;
            bf16_t v0 = (bf16_t)h0[r];
            bf16_t v1 = (bf16_t)h1[r];
            hbuf[row * E_ + c0 + n] = v0;
            hbuf[row * E_ + c1 + n] = v1;
            size_t go = (((size_t)(bbase + row)) * T_ + t) * E_;
            seq[go + c0 + n] = v0;
            seq[go + c1 + n] = v1;
        }
        __syncthreads();
    }

    #pragma unroll
    for (int r = 0; r < 8; ++r) {
        int row = r + half * 8;
        size_t go = (size_t)(bbase + row) * E_;
        hstate[go + c0 + n] = (bf16_t)h0[r];
        hstate[go + c1 + n] = (bf16_t)h1[r];
    }
}

// ---------------------------------------------------------------------------
// Layers 1..2: per step, acc = b + x@W + h@U with BOTH W and U B-fragments
// register-resident across the whole scan (256 VGPRs of weights/wave).
// seq_prev[:,t+1,:] is DOUBLE-BUFFER prefetched: global loads issue at the
// top of step t, the LDS stores land before the pre-write barrier, so the
// global round-trip is hidden behind the step's 32 WMMAs.
// seq is updated in place (data for step t was staged during step t-1).
// ---------------------------------------------------------------------------
__global__ __launch_bounds__(256, 1)
void rnn_layerN_kernel(const int*   __restrict__ nargs,  // [B]
                       const float* __restrict__ bias,   // [E] f32
                       const bf16_t* __restrict__ Wt,    // [E][E] bf16
                       const bf16_t* __restrict__ Ut,    // [E][E] bf16
                       bf16_t* __restrict__ seq,         // [B][T][E] (in/out)
                       bf16_t* __restrict__ hstate) {    // [B][E]    (in/out)
    __shared__ __align__(16) bf16_t hbuf[16 * E_];       // 8 KB
    __shared__ __align__(16) bf16_t xb[2][16 * E_];      // 16 KB double buffer
    __shared__ int snargs[16];

    const int tid   = threadIdx.x;
    const int lane  = tid & 31;
    const int wave  = tid >> 5;
    const int half  = lane >> 4;
    const int n     = lane & 15;
    const int bbase = blockIdx.x * 16;

    // staging coordinates: each thread owns 16 bf16 (32 B) of the 16x256 slab
    const int srow = tid >> 4;
    const int sc16 = (tid & 15) * 16;
    const bf16_t* sbase =
        seq + (((size_t)(bbase + srow)) * T_) * E_ + sc16;   // + t*E_ per step

    if (tid < 16) snargs[tid] = nargs[bbase + tid];

    {   // init h0 from previous layer's last state, and stage t=0 input
        const bf16_t* src = hstate + (size_t)(bbase + srow) * E_ + sc16;
        *(v8bf*)&hbuf[srow * E_ + sc16]     = *(const v8bf*)(src);
        *(v8bf*)&hbuf[srow * E_ + sc16 + 8] = *(const v8bf*)(src + 8);
        *(v8bf*)&xb[0][srow * E_ + sc16]     = *(const v8bf*)(sbase);
        *(v8bf*)&xb[0][srow * E_ + sc16 + 8] = *(const v8bf*)(sbase + 8);
    }
    __syncthreads();

    const int c0 = wave * 32;
    const int c1 = c0 + 16;
    const float bv0 = bias[c0 + n];
    const float bv1 = bias[c1 + n];

    int na[8];
    #pragma unroll
    for (int r = 0; r < 8; ++r) na[r] = snargs[r + half * 8];

    v8f h0, h1;
    #pragma unroll
    for (int r = 0; r < 8; ++r) {
        int row = r + half * 8;
        h0[r] = (float)hbuf[row * E_ + c0 + n];
        h1[r] = (float)hbuf[row * E_ + c1 + n];
    }

    // hoist ALL weight fragments for the scan (W and U, 256 VGPRs)
    v16bf wf0[8], wf1[8], uf0[8], uf1[8];
    #pragma unroll
    for (int ks = 0; ks < 8; ++ks) {
        int k0 = ks * 32;
        wf0[ks] = load_b_frag(Wt, E_, c0 + n, half, k0);
        wf1[ks] = load_b_frag(Wt, E_, c1 + n, half, k0);
        uf0[ks] = load_b_frag(Ut, E_, c0 + n, half, k0);
        uf1[ks] = load_b_frag(Ut, E_, c1 + n, half, k0);
    }

    for (int t = 0; t < T_; ++t) {
        const bf16_t* xcur = xb[t & 1];
        bf16_t*       xnxt = xb[(t + 1) & 1];

        // prefetch next step's input slab into registers (issues early,
        // overlaps with this step's WMMA chain)
        v8bf p0, p1;
        if (t + 1 < T_) {
            const bf16_t* src = sbase + (size_t)(t + 1) * E_;
            p0 = *(const v8bf*)(src);
            p1 = *(const v8bf*)(src + 8);
        }

        v8f acc0, acc1;
        #pragma unroll
        for (int r = 0; r < 8; ++r) { acc0[r] = bv0; acc1[r] = bv1; }

        #pragma unroll
        for (int ks = 0; ks < 8; ++ks) {
            v16bf a = load_a_frag(xcur, E_, n, half, ks * 32);
            acc0 = wmma_bf16(a, wf0[ks], acc0);
            acc1 = wmma_bf16(a, wf1[ks], acc1);
        }
        #pragma unroll
        for (int ks = 0; ks < 8; ++ks) {
            v16bf a = load_a_frag(hbuf, E_, n, half, ks * 32);
            acc0 = wmma_bf16(a, uf0[ks], acc0);
            acc1 = wmma_bf16(a, uf1[ks], acc1);
        }

        #pragma unroll
        for (int r = 0; r < 8; ++r) {
            bool mt = t < na[r];
            h0[r] = mt ? fast_tanh(acc0[r]) : h0[r];
            h1[r] = mt ? fast_tanh(acc1[r]) : h1[r];
        }

        // land the prefetched slab in the inactive LDS buffer
        if (t + 1 < T_) {
            *(v8bf*)&xnxt[srow * E_ + sc16]     = p0;
            *(v8bf*)&xnxt[srow * E_ + sc16 + 8] = p1;
        }

        __syncthreads();   // all waves done reading hbuf/xcur

        #pragma unroll
        for (int r = 0; r < 8; ++r) {
            int row = r + half * 8;
            bf16_t v0 = (bf16_t)h0[r];
            bf16_t v1 = (bf16_t)h1[r];
            hbuf[row * E_ + c0 + n] = v0;
            hbuf[row * E_ + c1 + n] = v1;
            size_t go = (((size_t)(bbase + row)) * T_ + t) * E_;
            seq[go + c0 + n] = v0;
            seq[go + c1 + n] = v1;
        }
        __syncthreads();   // hbuf + staged xnxt visible for next step
    }

    #pragma unroll
    for (int r = 0; r < 8; ++r) {
        int row = r + half * 8;
        size_t go = (size_t)(bbase + row) * E_;
        hstate[go + c0 + n] = (bf16_t)h0[r];
        hstate[go + c1 + n] = (bf16_t)h1[r];
    }
}

// ---------------------------------------------------------------------------
// Final projection: out = tanh(seq2 @ Wf + bf) * mask   ([B*T,256]x[256,512])
// ---------------------------------------------------------------------------
__global__ __launch_bounds__(256)
void final_proj_kernel(const bf16_t* __restrict__ seq,   // [B][T][E] bf16
                       const bf16_t* __restrict__ Wft,   // [H][E] bf16
                       const float*  __restrict__ bf_,   // [H] f32
                       const int*    __restrict__ nargs, // [B]
                       float* __restrict__ out) {        // [B][T][H] f32
    __shared__ __align__(16) bf16_t abuf[16 * E_];

    const int tid  = threadIdx.x;
    const int lane = tid & 31;
    const int wave = tid >> 5;
    const int half = lane >> 4;
    const int n    = lane & 15;
    const int b    = blockIdx.x;
    const int nsl  = blockIdx.y * 256;

    {
        const int row = tid >> 4;
        const int c16 = (tid & 15) * 16;
        const bf16_t* src = seq + (((size_t)b) * T_ + row) * E_ + c16;
        *(v8bf*)&abuf[row * E_ + c16]     = *(const v8bf*)(src);
        *(v8bf*)&abuf[row * E_ + c16 + 8] = *(const v8bf*)(src + 8);
    }
    __syncthreads();

    const int numa = nargs[b];
    const int c0 = nsl + wave * 32;
    const int c1 = c0 + 16;
    const float bv0 = bf_[c0 + n];
    const float bv1 = bf_[c1 + n];

    v8f acc0, acc1;
    #pragma unroll
    for (int r = 0; r < 8; ++r) { acc0[r] = bv0; acc1[r] = bv1; }

    #pragma unroll
    for (int ks = 0; ks < E_ / 32; ++ks) {
        int k0 = ks * 32;
        v16bf a  = load_a_frag(abuf, E_, n, half, k0);
        v16bf w0 = load_b_frag(Wft, E_, c0 + n, half, k0);
        v16bf w1 = load_b_frag(Wft, E_, c1 + n, half, k0);
        acc0 = wmma_bf16(a, w0, acc0);
        acc1 = wmma_bf16(a, w1, acc1);
    }

    #pragma unroll
    for (int r = 0; r < 8; ++r) {
        int trow = r + half * 8;
        float m = (trow < numa) ? 1.0f : 0.0f;
        size_t go = (((size_t)b) * T_ + trow) * (size_t)H_;
        out[go + c0 + n] = m * fast_tanh(acc0[r]);
        out[go + c1 + n] = m * fast_tanh(acc1[r]);
    }
}

// ---------------------------------------------------------------------------
// Launch
// ---------------------------------------------------------------------------
extern "C" void kernel_launch(void* const* d_in, const int* in_sizes, int n_in,
                              void* d_out, int out_size, void* d_ws, size_t ws_size,
                              hipStream_t stream) {
    const float* hs    = (const float*)d_in[0];   // [B][H]
    const float* temb  = (const float*)d_in[1];   // [B][E]
    const int*   nargs = (const int*)  d_in[2];   // [B]
    const float* W0 = (const float*)d_in[3];
    const float* U0 = (const float*)d_in[4];
    const float* b0 = (const float*)d_in[5];
    const float* W1 = (const float*)d_in[6];
    const float* U1 = (const float*)d_in[7];
    const float* b1 = (const float*)d_in[8];
    const float* W2 = (const float*)d_in[9];
    const float* U2 = (const float*)d_in[10];
    const float* b2 = (const float*)d_in[11];
    const float* Wf = (const float*)d_in[12];
    const float* bf = (const float*)d_in[13];
    float* out = (float*)d_out;

    // workspace layout (bf16)
    bf16_t* seq    = (bf16_t*)d_ws;                        // B*T*E   (32 MB)
    bf16_t* hstate = seq    + (size_t)B_ * T_ * E_;        // B*E     (2 MB)
    bf16_t* W0t    = hstate + (size_t)B_ * E_;             // [E][H]
    bf16_t* U0t    = W0t + (size_t)E_ * H_;                // [E][E]
    bf16_t* W1t    = U0t + (size_t)E_ * E_;
    bf16_t* U1t    = W1t + (size_t)E_ * E_;
    bf16_t* W2t    = U1t + (size_t)E_ * E_;
    bf16_t* U2t    = W2t + (size_t)E_ * E_;
    bf16_t* Wft    = U2t + (size_t)E_ * E_;                // [H][E]

    // weight prep: transpose + bf16 convert
    transpose_to_bf16<<<(H_ * E_ + 255) / 256, 256, 0, stream>>>(W0, W0t, H_, E_);
    transpose_to_bf16<<<(E_ * E_ + 255) / 256, 256, 0, stream>>>(U0, U0t, E_, E_);
    transpose_to_bf16<<<(E_ * E_ + 255) / 256, 256, 0, stream>>>(W1, W1t, E_, E_);
    transpose_to_bf16<<<(E_ * E_ + 255) / 256, 256, 0, stream>>>(U1, U1t, E_, E_);
    transpose_to_bf16<<<(E_ * E_ + 255) / 256, 256, 0, stream>>>(W2, W2t, E_, E_);
    transpose_to_bf16<<<(E_ * E_ + 255) / 256, 256, 0, stream>>>(U2, U2t, E_, E_);
    transpose_to_bf16<<<(E_ * H_ + 255) / 256, 256, 0, stream>>>(Wf, Wft, E_, H_);

    // three stacked RNN layers (state chained through hstate, seq in place)
    rnn_layer0_kernel<<<B_ / 16, 256, 0, stream>>>(hs, temb, nargs, b0, W0t, U0t,
                                                   seq, hstate);
    rnn_layerN_kernel<<<B_ / 16, 256, 0, stream>>>(nargs, b1, W1t, U1t, seq, hstate);
    rnn_layerN_kernel<<<B_ / 16, 256, 0, stream>>>(nargs, b2, W2t, U2t, seq, hstate);

    // final projection to output
    final_proj_kernel<<<dim3(B_, 2), 256, 0, stream>>>(seq, Wft, bf, nargs, out);
}
